// Seq2Seq_37245956391409
// MI455X (gfx1250) — matrix operand order repaired and checked by reading, not compiled, not run on hardware
//
#include <hip/hip_runtime.h>
#include <hip/hip_bf16.h>
#include <math.h>

#define B_  32
#define S_  64
#define T_  64
#define V_  32000
#define E_  512
#define H_  1024
#define G4H (4 * H_)

typedef __attribute__((ext_vector_type(16))) __bf16 v16bf;
typedef __attribute__((ext_vector_type(8)))  float  v8f;
typedef unsigned short u16;
typedef unsigned int   u32;

// ---------- helpers ----------
__device__ __forceinline__ u16 f2bf(float f) {
  u32 u = __float_as_uint(f);
  u32 r = (u + 0x7FFFu + ((u >> 16) & 1u)) >> 16;   // round-to-nearest-even
  return (u16)r;
}
__device__ __forceinline__ float sigm(float x) { return 1.f / (1.f + __expf(-x)); }

// ---------- weight conversion fp32 -> bf16 ----------
__global__ void k_f32_to_bf16(const float* __restrict__ in, u16* __restrict__ out, int n) {
  int i = blockIdx.x * blockDim.x + threadIdx.x;
  if (i < n) out[i] = f2bf(in[i]);
}

__global__ void k_zero32(u32* __restrict__ p, int n) {
  int i = blockIdx.x * blockDim.x + threadIdx.x;
  if (i < n) p[i] = 0u;
}

// encoder embeddings: xs[(s*B+b)*E + e] = (tok==0 ? 0 : tab[tok][e]), tok = src[b][s]
__global__ void k_embed_enc(const int* __restrict__ src, const float* __restrict__ tab,
                            u16* __restrict__ xs) {
  int i = blockIdx.x * blockDim.x + threadIdx.x;
  if (i >= S_ * B_ * E_) return;
  int e  = i % E_;
  int sb = i / E_;
  int b  = sb % B_;
  int s  = sb / B_;
  int tok = src[b * S_ + s];
  float v = (tok == 0) ? 0.f : tab[(long)tok * E_ + e];
  xs[i] = f2bf(v);
}

// decoder embedding of current tokens: x[b][e]
__global__ void k_embed_dec(const int* __restrict__ tok, const float* __restrict__ tab,
                            u16* __restrict__ x) {
  int i = blockIdx.x * blockDim.x + threadIdx.x;
  if (i >= B_ * E_) return;
  int e = i % E_;
  int b = i / E_;
  int tk = tok[b];
  float v = (tk == 0) ? 0.f : tab[(long)tk * E_ + e];
  x[i] = f2bf(v);
}

// ---------- WMMA fragment loads ----------
// A (16x32 bf16, rows m0..m0+15): lane m = lane&15, split-8 K chunks per ISA layout
__device__ __forceinline__ v16bf load_fragA(const u16* __restrict__ A, int K,
                                            int m0, int k0, int lane) {
  int m = m0 + (lane & 15);
  const u16* p = A + (long)m * K + k0 + ((lane >> 4) * 8);
  union { v16bf v; uint4 q[2]; } u;
  u.q[0] = *(const uint4*)(p);        // K = k0+koff .. +7
  u.q[1] = *(const uint4*)(p + 16);   // K = k0+16+koff .. +7
  return u.v;
}
// B (32x16 = KxN): weights stored row-major (N,K); lane n = lane&15,
// split-8 K chunks matching A: K = k0+koff..+7 and k0+16+koff..+7, koff = (lane>>4)*8
__device__ __forceinline__ v16bf load_fragB(const u16* __restrict__ W, int K,
                                            int n0, int k0, int lane) {
  int n = n0 + (lane & 15);
  const u16* p = W + (long)n * K + k0 + ((lane >> 4) * 8);
  union { v16bf v; uint4 q[2]; } u;
  u.q[0] = *(const uint4*)(p);
  u.q[1] = *(const uint4*)(p + 16);
  return u.v;
}

#define WMMA_BF16(a, b, c) \
  __builtin_amdgcn_wmma_f32_16x16x32_bf16(false, (a), false, (b), (short)0, (c), false, false)

// One K-segment of C += A * W^T for a 32x32 output tile (2 M-tiles x 2 N-tiles),
// software-pipelined: next chunk's fragments are loaded before current WMMAs issue.
__device__ __forceinline__ void gemm_seg(const u16* __restrict__ A, const u16* __restrict__ W,
                                         int K, int n0, int lane,
                                         v8f& acc00, v8f& acc10, v8f& acc01, v8f& acc11) {
  v16bf a0 = load_fragA(A, K, 0,  0, lane);
  v16bf a1 = load_fragA(A, K, 16, 0, lane);
  v16bf b0 = load_fragB(W, K, n0,      0, lane);
  v16bf b1 = load_fragB(W, K, n0 + 16, 0, lane);
  for (int k0 = 32; k0 < K; k0 += 32) {
    v16bf na0 = load_fragA(A, K, 0,  k0, lane);
    v16bf na1 = load_fragA(A, K, 16, k0, lane);
    v16bf nb0 = load_fragB(W, K, n0,      k0, lane);
    v16bf nb1 = load_fragB(W, K, n0 + 16, k0, lane);
    if (k0 + 32 < K)
      __builtin_prefetch(W + (long)(n0 + (lane & 15)) * K + k0 + 32, 0, 1);
    acc00 = WMMA_BF16(a0, b0, acc00);
    acc10 = WMMA_BF16(a1, b0, acc10);
    acc01 = WMMA_BF16(a0, b1, acc01);
    acc11 = WMMA_BF16(a1, b1, acc11);
    a0 = na0; a1 = na1; b0 = nb0; b1 = nb1;
  }
  acc00 = WMMA_BF16(a0, b0, acc00);
  acc10 = WMMA_BF16(a1, b0, acc10);
  acc01 = WMMA_BF16(a0, b1, acc01);
  acc11 = WMMA_BF16(a1, b1, acc11);
}

// out(32 x N) = A1(32xK1)*W1(NxK1)^T [+ A2(32xK2)*W2(NxK2)^T] + bias, fp32 out.
// One wave per 32-wide N tile (two 16-wide WMMA N tiles), full M=32.
__global__ void k_gemm_dual_wmma(const u16* __restrict__ A1, const u16* __restrict__ W1, int K1,
                                 const u16* __restrict__ A2, const u16* __restrict__ W2, int K2,
                                 const float* __restrict__ bias,
                                 float* __restrict__ out, long ostride, int NT32) {
  int wave = (int)((blockIdx.x * blockDim.x + threadIdx.x) >> 5);
  int lane = threadIdx.x & 31;
  if (wave >= NT32) return;               // wave-uniform: EXEC stays all-ones for WMMA
  int n0 = wave * 32;

  v8f acc00 = {};   // rows 0-15,  cols n0..n0+15
  v8f acc10 = {};   // rows 16-31, cols n0..n0+15
  v8f acc01 = {};   // rows 0-15,  cols n0+16..n0+31
  v8f acc11 = {};   // rows 16-31, cols n0+16..n0+31

  gemm_seg(A1, W1, K1, n0, lane, acc00, acc10, acc01, acc11);
  if (K2 > 0)
    gemm_seg(A2, W2, K2, n0, lane, acc00, acc10, acc01, acc11);

  int nc = lane & 15;
  int mb = (lane >> 4) * 8;               // C/D layout: VGPR r = row mb+r
  float bv0 = bias[n0 + nc];
  float bv1 = bias[n0 + 16 + nc];
#pragma unroll
  for (int r = 0; r < 8; ++r) {
    out[(long)(mb + r)      * ostride + n0 + nc]      = acc00[r] + bv0;
    out[(long)(mb + 16 + r) * ostride + n0 + nc]      = acc10[r] + bv0;
    out[(long)(mb + r)      * ostride + n0 + 16 + nc] = acc01[r] + bv1;
    out[(long)(mb + 16 + r) * ostride + n0 + 16 + nc] = acc11[r] + bv1;
  }
}

// LSTM gate nonlinearity + state update. g layout (B,4H) = [i|f|g|o].
__global__ void k_lstm_gates(const float* __restrict__ g, float* __restrict__ c,
                             u16* __restrict__ h_bf, u16* __restrict__ ys /* nullable */) {
  int i = blockIdx.x * blockDim.x + threadIdx.x;
  if (i >= B_ * H_) return;
  int j = i % H_;
  int b = i / H_;
  const float* gr = g + (long)b * G4H;
  float gi = gr[j];
  float gf = gr[H_ + j];
  float gc = gr[2 * H_ + j];
  float go = gr[3 * H_ + j];
  float cn = sigm(gf) * c[i] + sigm(gi) * tanhf(gc);
  float hn = sigm(go) * tanhf(cn);
  c[i] = cn;
  u16 hb = f2bf(hn);
  h_bf[i] = hb;
  if (ys) ys[i] = hb;
}

// argmax over V + teacher forcing select
__global__ void k_argmax_select(const float* __restrict__ logits, long rstride,
                                const int* __restrict__ tgt, const int* __restrict__ tmask,
                                int t, int* __restrict__ nxt) {
  __shared__ float sv[256];
  __shared__ int   si[256];
  int b = blockIdx.x;
  int tid = threadIdx.x;
  const float* row = logits + (long)b * rstride;
  float best = -3.4e38f;
  int   bi   = 0;
  for (int v = tid; v < V_; v += 256) {
    float x = row[v];
    if (x > best) { best = x; bi = v; }
  }
  sv[tid] = best; si[tid] = bi;
  __syncthreads();
  for (int off = 128; off > 0; off >>= 1) {
    if (tid < off) {
      float xo = sv[tid + off]; int io = si[tid + off];
      if (xo > sv[tid] || (xo == sv[tid] && io < si[tid])) { sv[tid] = xo; si[tid] = io; }
    }
    __syncthreads();
  }
  if (tid == 0) nxt[b] = (tmask[t] > 0) ? tgt[b * T_ + t] : si[0];
}

__global__ void k_init_tok(const int* __restrict__ tgt, int* __restrict__ nxt) {
  int b = threadIdx.x;
  if (b < B_) nxt[b] = tgt[b * T_];
}

__global__ void k_zero_t0(float* __restrict__ out) {
  int i = blockIdx.x * blockDim.x + threadIdx.x;
  if (i >= B_ * V_) return;
  int b = i / V_;
  int v = i % V_;
  out[(long)b * ((long)T_ * V_) + v] = 0.f;
}

// ---------- host ----------
extern "C" void kernel_launch(void* const* d_in, const int* in_sizes, int n_in,
                              void* d_out, int out_size, void* d_ws, size_t ws_size,
                              hipStream_t stream) {
  (void)in_sizes; (void)n_in; (void)out_size; (void)ws_size;

  const int*   src       = (const int*)  d_in[0];
  const int*   tgt       = (const int*)  d_in[1];
  const int*   tmask     = (const int*)  d_in[2];
  const float* enc_embed = (const float*)d_in[3];
  const float* dec_embed = (const float*)d_in[4];
  const float* enc_wih0  = (const float*)d_in[5];
  const float* enc_whh0  = (const float*)d_in[6];
  const float* enc_b0    = (const float*)d_in[7];
  const float* enc_wih1  = (const float*)d_in[8];
  const float* enc_whh1  = (const float*)d_in[9];
  const float* enc_b1    = (const float*)d_in[10];
  const float* dec_wih0  = (const float*)d_in[11];
  const float* dec_whh0  = (const float*)d_in[12];
  const float* dec_b0    = (const float*)d_in[13];
  const float* dec_wih1  = (const float*)d_in[14];
  const float* dec_whh1  = (const float*)d_in[15];
  const float* dec_b1    = (const float*)d_in[16];
  const float* fc_w      = (const float*)d_in[17];
  const float* fc_b      = (const float*)d_in[18];
  float* out = (float*)d_out;

  char* ws = (char*)d_ws;
  size_t off = 0;
  auto alloc = [&](size_t bytes) -> char* {
    char* p = ws + off;
    off += (bytes + 255) & ~(size_t)255;
    return p;
  };

  u16* w_ewih0 = (u16*)alloc((size_t)G4H * E_ * 2);
  u16* w_ewhh0 = (u16*)alloc((size_t)G4H * H_ * 2);
  u16* w_ewih1 = (u16*)alloc((size_t)G4H * H_ * 2);
  u16* w_ewhh1 = (u16*)alloc((size_t)G4H * H_ * 2);
  u16* w_dwih0 = (u16*)alloc((size_t)G4H * E_ * 2);
  u16* w_dwhh0 = (u16*)alloc((size_t)G4H * H_ * 2);
  u16* w_dwih1 = (u16*)alloc((size_t)G4H * H_ * 2);
  u16* w_dwhh1 = (u16*)alloc((size_t)G4H * H_ * 2);
  u16* w_fc    = (u16*)alloc((size_t)V_  * H_ * 2);
  u16* xs      = (u16*)alloc((size_t)S_ * B_ * E_ * 2);
  u16* ys0     = (u16*)alloc((size_t)S_ * B_ * H_ * 2);
  float* cA    = (float*)alloc((size_t)B_ * H_ * 4);
  float* cB    = (float*)alloc((size_t)B_ * H_ * 4);
  u16* hA      = (u16*)alloc((size_t)B_ * H_ * 2);
  u16* hB      = (u16*)alloc((size_t)B_ * H_ * 2);
  u16* xdec    = (u16*)alloc((size_t)B_ * E_ * 2);
  float* gbuf  = (float*)alloc((size_t)B_ * G4H * 4);
  int* nxt     = (int*)alloc((size_t)B_ * 4);

  auto conv = [&](const float* s, u16* d, int n) {
    k_f32_to_bf16<<<(n + 255) / 256, 256, 0, stream>>>(s, d, n);
  };
  conv(enc_wih0, w_ewih0, G4H * E_);
  conv(enc_whh0, w_ewhh0, G4H * H_);
  conv(enc_wih1, w_ewih1, G4H * H_);
  conv(enc_whh1, w_ewhh1, G4H * H_);
  conv(dec_wih0, w_dwih0, G4H * E_);
  conv(dec_whh0, w_dwhh0, G4H * H_);
  conv(dec_wih1, w_dwih1, G4H * H_);
  conv(dec_whh1, w_dwhh1, G4H * H_);
  conv(fc_w,     w_fc,    V_ * H_);

  // zero states (c fp32, h bf16)
  k_zero32<<<(B_ * H_ + 255) / 256, 256, 0, stream>>>((u32*)cA, B_ * H_);
  k_zero32<<<(B_ * H_ + 255) / 256, 256, 0, stream>>>((u32*)cB, B_ * H_);
  k_zero32<<<(B_ * H_ / 2 + 255) / 256, 256, 0, stream>>>((u32*)hA, B_ * H_ / 2);
  k_zero32<<<(B_ * H_ / 2 + 255) / 256, 256, 0, stream>>>((u32*)hB, B_ * H_ / 2);

  // encoder embeddings
  {
    int n = S_ * B_ * E_;
    k_embed_enc<<<(n + 255) / 256, 256, 0, stream>>>(src, enc_embed, xs);
  }

  auto gemm = [&](const u16* A1, const u16* W1, int K1,
                  const u16* A2, const u16* W2, int K2,
                  const float* bias, float* o, long ostride, int N) {
    int NT32 = N / 32;                 // one wave per 32-wide N tile
    int grid = (NT32 + 3) / 4;         // 4 waves / 128-thread block
    k_gemm_dual_wmma<<<grid, 128, 0, stream>>>(A1, W1, K1, A2, W2, K2, bias, o, ostride, NT32);
  };
  auto gates = [&](float* g, float* c, u16* h, u16* ys) {
    k_lstm_gates<<<(B_ * H_ + 255) / 256, 256, 0, stream>>>(g, c, h, ys);
  };

  // ---- encoder layer 0 ----
  for (int s = 0; s < S_; ++s) {
    gemm(xs + (long)s * B_ * E_, w_ewih0, E_, hA, w_ewhh0, H_, enc_b0, gbuf, G4H, G4H);
    gates(gbuf, cA, hA, ys0 + (long)s * B_ * H_);
  }
  // ---- encoder layer 1 ----
  for (int s = 0; s < S_; ++s) {
    gemm(ys0 + (long)s * B_ * H_, w_ewih1, H_, hB, w_ewhh1, H_, enc_b1, gbuf, G4H, G4H);
    gates(gbuf, cB, hB, nullptr);
  }

  // ---- decoder ----
  k_init_tok<<<1, 32, 0, stream>>>(tgt, nxt);
  k_zero_t0<<<(B_ * V_ + 255) / 256, 256, 0, stream>>>(out);

  for (int t = 1; t < T_; ++t) {
    k_embed_dec<<<(B_ * E_ + 255) / 256, 256, 0, stream>>>(nxt, dec_embed, xdec);
    gemm(xdec, w_dwih0, E_, hA, w_dwhh0, H_, dec_b0, gbuf, G4H, G4H);
    gates(gbuf, cA, hA, nullptr);
    gemm(hA, w_dwih1, H_, hB, w_dwhh1, H_, dec_b1, gbuf, G4H, G4H);
    gates(gbuf, cB, hB, nullptr);
    // logits -> d_out[:, t, :]
    gemm(hB, w_fc, H_, nullptr, nullptr, 0, fc_b, out + (long)t * V_, (long)T_ * V_, V_);
    k_argmax_select<<<B_, 256, 0, stream>>>(out + (long)t * V_, (long)T_ * V_, tgt, tmask, t, nxt);
  }
}
